// CSPN_83605833384072
// MI455X (gfx1250) — matrix operand order repaired and checked by reading, not compiled, not run on hardware
//
#include <hip/hip_runtime.h>
#include <hip/hip_bf16.h>
#include <stdint.h>
#include <stddef.h>

typedef __attribute__((ext_vector_type(16))) _Float16 v16h;
typedef __attribute__((ext_vector_type(8)))  float    v8f;
typedef __attribute__((__vector_size__(16))) int      v4i_;

#if defined(__gfx1250__) && __has_builtin(__builtin_amdgcn_global_load_async_to_lds_b128)
#define HAVE_ASYNC_LDS 1
#endif

namespace {
constexpr int cB  = 4;
constexpr int cC  = 64;
constexpr int cH  = 256;
constexpr int cW  = 320;
constexpr int cHW = cH * cW;            // 81920
constexpr int cNP = cB * cHW;           // 327680 pixels total
constexpr int cNCONV = 83;              // 8 (w3) + 24 (w5) + 48 (w7) + 3 (ck)
constexpr int cKST = 18;                // K = 576 = 18 * 32
constexpr int cMT  = 6;                 // 96 GEMM rows (83 used)
}

// ---------------------------------------------------------------------------
// Kernel 0: fout f32 NCHW -> f16 NHWC (channel-innermost) via LDS transpose.
// One block = one (b, y, 64-pixel strip). Output pixel record = 128 contiguous
// bytes so downstream staging can use 16B vector / async-LDS transfers.
// ---------------------------------------------------------------------------
__global__ __launch_bounds__(256) void cvt_nhwc(const float* __restrict__ in,
                                                _Float16* __restrict__ out) {
  __shared__ _Float16 s[64 * 65];            // [c][x], padded to kill conflicts
  const int t  = threadIdx.x;
  const int x0 = blockIdx.x * 64;
  const int y  = blockIdx.y;
  const int b  = blockIdx.z;

  const int x  = t & 63;
  const int c0 = t >> 6;                     // 0..3
  for (int c = c0; c < 64; c += 4)
    s[c * 65 + x] = (_Float16)in[((size_t)(b * cC + c) * cH + y) * cW + x0 + x];
  __syncthreads();

  const int px = t >> 2;                     // pixel within strip
  const int g  = t & 3;                      // 16-channel group
  v16h v;
  #pragma unroll
  for (int j = 0; j < 16; ++j) v[j] = s[(g * 16 + j) * 65 + px];
  *(v16h*)(out + (((size_t)(b * cH + y) * cW + x0 + px) * 64 + g * 16)) = v;
}

// ---------------------------------------------------------------------------
// Kernel 1: pack conv weights (w3|w5|w7|ck) into WMMA A-fragment layout (f16).
// Global K order: k = (ky*3+kx)*64 + c.
// A 16x32 lane layout (matches contiguous B): lanes 0-15 M=lane, K=e;
// lanes 16-31 M=lane-16, K=16+e.
// frag half-index = (((mt*18)+ks)*32 + lane)*16 + e
// ---------------------------------------------------------------------------
__global__ __launch_bounds__(256) void prep_afrag(const float* __restrict__ w3,
                                                  const float* __restrict__ w5,
                                                  const float* __restrict__ w7,
                                                  const float* __restrict__ ck,
                                                  _Float16* __restrict__ frag) {
  int idx = blockIdx.x * blockDim.x + threadIdx.x;
  const int total = cMT * cKST * 32 * 16;             // 55296
  if (idx >= total) return;
  const int e  = idx & 15;
  const int l  = (idx >> 4) & 31;
  const int ks = (idx >> 9) % cKST;
  const int mt = idx / (cKST * 512);
  const int baseLo = (l & 16) >> 1;                   // 0 or 8
  const int ke = (baseLo << 1) + e;
  const int k  = ks * 32 + ke;                        // 0..575
  const int t  = k >> 6;                              // tap ky*3+kx
  const int c  = k & 63;
  const int ky = t / 3, kx = t % 3;
  const int m  = mt * 16 + (l & 15);
  float v = 0.f;
  if (m < 8)        v = w3[((m       * 64 + c) * 3 + ky) * 3 + kx];
  else if (m < 32)  v = w5[(((m - 8) * 64 + c) * 3 + ky) * 3 + kx];
  else if (m < 80)  v = w7[(((m - 32)* 64 + c) * 3 + ky) * 3 + kx];
  else if (m < 83)  v = ck[(((m - 80)* 64 + c) * 3 + ky) * 3 + kx];
  frag[idx] = (_Float16)v;
}

// ---------------------------------------------------------------------------
// Kernel 2: implicit-GEMM 3x3 SAME conv via WMMA f16 -> f32.
// Workgroup = one 64-pixel row strip x 96 channels. 8 waves:
//   wave = 4*mhalf + ncol ; each wave owns 3 M-tiles x 1 N-column (16 px).
// LDS input tile: [ky 0..2][col 0..65][c 0..63] f16 (25,344 B); a lane's
// B-fragment (16 contiguous channels) is one aligned 32-byte LDS load.
// NHWC input => each staged element is 16B contiguous; staged with
// GLOBAL_LOAD_ASYNC_TO_LDS_B128 (ASYNCcnt) when the toolchain exposes it.
// ---------------------------------------------------------------------------
__global__ __launch_bounds__(256) void conv_wmma(const _Float16* __restrict__ in16,
                                                 const v16h* __restrict__ afrag,
                                                 float* __restrict__ convout) {
  __shared__ _Float16 sIn[3 * 66 * 64];
  const int tid = threadIdx.x;
  const int x0  = blockIdx.x * 64;           // 0,64,...,256
  const int y   = blockIdx.y;                // 0..255
  const int b   = blockIdx.z;                // 0..3

  // Boundary tiles need zero padding in LDS (async loads can't write zeros).
  const bool boundary = (y == 0) || (y == cH - 1) || (x0 == 0) || (x0 + 64 >= cW);
  if (boundary) {
    const v16h z = {};
    for (int i = tid; i < (3 * 66 * 64) / 16; i += 256) ((v16h*)sIn)[i] = z;
  }
  __syncthreads();

  // 198 (ky,col) pairs x 8 chunks of 16B each.
  for (int idx = tid; idx < 198 * 8; idx += 256) {
    const int pair = idx >> 3, q = idx & 7;
    const int ky = pair / 66, col = pair % 66;
    const int yy = y + ky - 1;
    const int xx = x0 + col - 1;
    if (yy >= 0 && yy < cH && xx >= 0 && xx < cW) {
      const _Float16* gsrc = in16 + (((size_t)(b * cH + yy) * cW + xx) * 64 + q * 8);
#ifdef HAVE_ASYNC_LDS
      __builtin_amdgcn_global_load_async_to_lds_b128(
          (__attribute__((address_space(1))) v4i_*)gsrc,
          (__attribute__((address_space(3))) v4i_*)&sIn[(pair << 6) + (q << 3)],
          0, 0);
#else
      ((uint4*)sIn)[idx] = *(const uint4*)gsrc;
#endif
    }
  }
#ifdef HAVE_ASYNC_LDS
#if __has_builtin(__builtin_amdgcn_s_wait_asynccnt)
  __builtin_amdgcn_s_wait_asynccnt(0);
#else
  asm volatile("s_wait_asynccnt 0x0" ::: "memory");
#endif
#endif
  __syncthreads();

  const int wave = tid >> 5, lane = tid & 31;
  const int ncol = wave & 3;                 // N column (16 px)
  const int mh   = wave >> 2;                // 0 -> M tiles 0..2, 1 -> 3..5
  const int px   = (ncol << 4) + (lane & 15);

  v8f acc0 = {}, acc1 = {}, acc2 = {};
  for (int ks = 0; ks < cKST; ++ks) {
    const int t  = ks >> 1;
    const int ky = t / 3, kx = t % 3;
    const int cb = ((ks & 1) << 5) | (lane & 16);   // contiguous channel base
    const v16h bf = *(const v16h*)&sIn[((ky * 66 + px + kx) << 6) + cb];
    const int mb = mh * 3;
    v16h a0 = afrag[((mb + 0) * cKST + ks) * 32 + lane];
    v16h a1 = afrag[((mb + 1) * cKST + ks) * 32 + lane];
    v16h a2 = afrag[((mb + 2) * cKST + ks) * 32 + lane];
    acc0 = __builtin_amdgcn_wmma_f32_16x16x32_f16(false, a0, false, bf, (short)0, acc0, false, false);
    acc1 = __builtin_amdgcn_wmma_f32_16x16x32_f16(false, a1, false, bf, (short)0, acc1, false, false);
    acc2 = __builtin_amdgcn_wmma_f32_16x16x32_f16(false, a2, false, bf, (short)0, acc2, false, false);
  }

  const int p = (b * cH + y) * cW + x0 + (ncol << 4) + (lane & 15);
  const int chAdd = (lane & 16) >> 1;        // upper half-lanes hold M+8
  #pragma unroll
  for (int r = 0; r < 8; ++r) {
    int ch = (mh * 3 + 0) * 16 + r + chAdd;
    if (ch < cNCONV) convout[(size_t)ch * cNP + p] = acc0[r];
    ch = (mh * 3 + 1) * 16 + r + chAdd;
    if (ch < cNCONV) convout[(size_t)ch * cNP + p] = acc1[r];
    ch = (mh * 3 + 2) * 16 + r + chAdd;
    if (ch < cNCONV) convout[(size_t)ch * cNP + p] = acc2[r];
  }
}

// ---------------------------------------------------------------------------
// Kernel 3a/3b: batchnorm stats, deterministic two-stage reduction.
// Stage A: 80 channels x 8 segments -> partial (sum, sumsq).
// Stage B: 80 threads fold the 8 partials in fixed order -> scale/shift.
// ---------------------------------------------------------------------------
__global__ __launch_bounds__(256) void bn_partial(const float* __restrict__ conv,
                                                  float* __restrict__ part) {
  __shared__ float ss[256], sq[256];
  const int ch = blockIdx.x, seg = blockIdx.y, tid = threadIdx.x;
  const int segN = cNP / 8;                       // 40960
  const float* src = conv + (size_t)ch * cNP + (size_t)seg * segN;
  float s = 0.f, q = 0.f;
  for (int i = tid; i < segN; i += 256) { const float v = src[i]; s += v; q += v * v; }
  ss[tid] = s; sq[tid] = q;
  __syncthreads();
  for (int off = 128; off > 0; off >>= 1) {
    if (tid < off) { ss[tid] += ss[tid + off]; sq[tid] += sq[tid + off]; }
    __syncthreads();
  }
  if (tid == 0) {
    part[(ch * 8 + seg) * 2 + 0] = ss[0];
    part[(ch * 8 + seg) * 2 + 1] = sq[0];
  }
}

__global__ __launch_bounds__(128) void bn_finalize(const float* __restrict__ part,
                                                   const float* __restrict__ g3, const float* __restrict__ b3,
                                                   const float* __restrict__ g5, const float* __restrict__ b5,
                                                   const float* __restrict__ g7, const float* __restrict__ b7,
                                                   float* __restrict__ scale, float* __restrict__ shift) {
  const int ch = threadIdx.x;
  if (ch >= 80) return;
  float s = 0.f, q = 0.f;
  for (int seg = 0; seg < 8; ++seg) {           // fixed order -> deterministic
    s += part[(ch * 8 + seg) * 2 + 0];
    q += part[(ch * 8 + seg) * 2 + 1];
  }
  const float inv = 1.f / (float)cNP;
  const float mean = s * inv;
  const float var  = q * inv - mean * mean;
  float g, bb;
  if (ch < 8)       { g = g3[ch];      bb = b3[ch];      }
  else if (ch < 32) { g = g5[ch - 8];  bb = b5[ch - 8];  }
  else              { g = g7[ch - 32]; bb = b7[ch - 32]; }
  const float sc = g * rsqrtf(var + 1e-5f);
  scale[ch] = sc;
  shift[ch] = bb - mean * sc;
}

// ---------------------------------------------------------------------------
// Kernel 4: per-pixel kernel generation (bn+relu+L1 norm+mid insert+gaussian+
// renorm) for d=3,5,7; confidence softmax; hns channel 0 = hn.
// ---------------------------------------------------------------------------
template <int D, int NCH>
__device__ __forceinline__ void gen_scale(const float* __restrict__ conv, int chBase,
                                          const float* __restrict__ scale,
                                          const float* __restrict__ shift,
                                          float sigma, float* __restrict__ kout, int p) {
  float raw[NCH];
  float s = 0.f;
  #pragma unroll
  for (int i = 0; i < NCH; ++i) {
    const int ch = chBase + i;
    float v = scale[ch] * conv[(size_t)ch * cNP + p] + shift[ch];
    v = v > 0.f ? v : 0.f;
    raw[i] = v; s += v;
  }
  const float inv = 1.f / (s + 1e-6f);
  const float mid = 1.f - s * inv;
  const float gden = 1.f / (2.f * sigma * sigma);
  constexpr int HB = NCH / 2;
  float tot = 0.f;
  #pragma unroll
  for (int j = 0; j <= NCH; ++j) {
    float wv = (j < HB) ? raw[j] * inv : ((j == HB) ? mid : raw[j - 1] * inv);
    const int iy = j / D, ix = j % D;
    const float ry = -1.f + 2.f * (float)iy / (float)(D - 1);
    const float rx = -1.f + 2.f * (float)ix / (float)(D - 1);
    tot += wv * __expf(-(rx * rx + ry * ry) * gden);
  }
  const float dn = 1.f / fmaxf(tot, 1e-7f);
  #pragma unroll
  for (int j = 0; j <= NCH; ++j) {
    float wv = (j < HB) ? raw[j] * inv : ((j == HB) ? mid : raw[j - 1] * inv);
    const int iy = j / D, ix = j % D;
    const float ry = -1.f + 2.f * (float)iy / (float)(D - 1);
    const float rx = -1.f + 2.f * (float)ix / (float)(D - 1);
    kout[(size_t)j * cNP + p] = wv * __expf(-(rx * rx + ry * ry) * gden) * dn;
  }
}

__global__ __launch_bounds__(256) void gen_kernels(const float* __restrict__ conv,
                                                   const float* __restrict__ scale,
                                                   const float* __restrict__ shift,
                                                   const float* __restrict__ sig3,
                                                   const float* __restrict__ sig5,
                                                   const float* __restrict__ sig7,
                                                   const float* __restrict__ ckb,
                                                   const float* __restrict__ hn,
                                                   float* __restrict__ kbuf,
                                                   float* __restrict__ conf,
                                                   float* __restrict__ hns) {
  const int p = blockIdx.x * blockDim.x + threadIdx.x;
  if (p >= cNP) return;
  gen_scale<3, 8>(conv, 0,  scale, shift, sig3[0], kbuf,                       p);
  gen_scale<5, 24>(conv, 8,  scale, shift, sig5[0], kbuf + (size_t)9  * cNP,   p);
  gen_scale<7, 48>(conv, 32, scale, shift, sig7[0], kbuf + (size_t)34 * cNP,   p);
  // confidence softmax over the 3 ck channels (conv rows 80..82) + bias
  float l0 = conv[(size_t)80 * cNP + p] + ckb[0];
  float l1 = conv[(size_t)81 * cNP + p] + ckb[1];
  float l2 = conv[(size_t)82 * cNP + p] + ckb[2];
  const float m = fmaxf(l0, fmaxf(l1, l2));
  const float e0 = __expf(l0 - m), e1 = __expf(l1 - m), e2 = __expf(l2 - m);
  const float is = 1.f / (e0 + e1 + e2);
  conf[p]                    = e0 * is;
  conf[(size_t)cNP + p]      = e1 * is;
  conf[(size_t)2 * cNP + p]  = e2 * is;
  hns[p] = hn[p];
}

// ---------------------------------------------------------------------------
// Kernel 5: one CSPN propagation step for all three scales (LDS halo tiles).
// comboCh >= 0 -> also write conf-weighted combination into hns[comboCh].
// ---------------------------------------------------------------------------
__global__ __launch_bounds__(256) void propagate(const float* __restrict__ s3,
                                                 const float* __restrict__ s5,
                                                 const float* __restrict__ s7,
                                                 const float* __restrict__ w3,
                                                 const float* __restrict__ w5,
                                                 const float* __restrict__ w7,
                                                 float* __restrict__ d3,
                                                 float* __restrict__ d5,
                                                 float* __restrict__ d7,
                                                 const float* __restrict__ conf,
                                                 float* __restrict__ hns, int comboCh) {
  __shared__ float t3[18 * 18];
  __shared__ float t5[20 * 20];
  __shared__ float t7[22 * 22];
  const int tid = threadIdx.x;
  const int gx0 = blockIdx.x * 16, gy0 = blockIdx.y * 16, b = blockIdx.z;
  const size_t base = (size_t)b * cHW;

  for (int i = tid; i < 18 * 18; i += 256) {
    const int ly = i / 18, lx = i % 18;
    const int gy = gy0 + ly - 1, gx = gx0 + lx - 1;
    t3[i] = (gy >= 0 && gy < cH && gx >= 0 && gx < cW) ? s3[base + gy * cW + gx] : 0.f;
  }
  for (int i = tid; i < 20 * 20; i += 256) {
    const int ly = i / 20, lx = i % 20;
    const int gy = gy0 + ly - 2, gx = gx0 + lx - 2;
    t5[i] = (gy >= 0 && gy < cH && gx >= 0 && gx < cW) ? s5[base + gy * cW + gx] : 0.f;
  }
  for (int i = tid; i < 22 * 22; i += 256) {
    const int ly = i / 22, lx = i % 22;
    const int gy = gy0 + ly - 3, gx = gx0 + lx - 3;
    t7[i] = (gy >= 0 && gy < cH && gx >= 0 && gx < cW) ? s7[base + gy * cW + gx] : 0.f;
  }
  __syncthreads();

  const int tx = tid & 15, ty = tid >> 4;
  const int p  = (b * cH + gy0 + ty) * cW + gx0 + tx;

  float a3 = 0.f;
  #pragma unroll
  for (int j = 0; j < 9; ++j)
    a3 += w3[(size_t)j * cNP + p] * t3[(ty + 1 + j / 3 - 1) * 18 + (tx + 1 + j % 3 - 1)];
  float a5 = 0.f;
  #pragma unroll
  for (int j = 0; j < 25; ++j)
    a5 += w5[(size_t)j * cNP + p] * t5[(ty + 2 + j / 5 - 2) * 20 + (tx + 2 + j % 5 - 2)];
  float a7 = 0.f;
  #pragma unroll
  for (int j = 0; j < 49; ++j)
    a7 += w7[(size_t)j * cNP + p] * t7[(ty + 3 + j / 7 - 3) * 22 + (tx + 3 + j % 7 - 3)];

  d3[p] = a3; d5[p] = a5; d7[p] = a7;
  if (comboCh >= 0) {
    hns[(size_t)comboCh * cNP + p] =
        conf[p] * a3 + conf[(size_t)cNP + p] * a5 + conf[(size_t)2 * cNP + p] * a7;
  }
}

// ---------------------------------------------------------------------------
// Kernel 6: final 67-ch 3x3 conv (ct), softmax over 3 outputs, blend with hns.
// ---------------------------------------------------------------------------
__global__ __launch_bounds__(256) void final_ct(const float* __restrict__ fout,
                                                const float* __restrict__ hns,
                                                const float* __restrict__ ctw,
                                                const float* __restrict__ ctb,
                                                float* __restrict__ out) {
  __shared__ float sw[3 * 67 * 9];
  __shared__ float sb[3];
  const int tid = threadIdx.x;
  for (int i = tid; i < 3 * 67 * 9; i += 256) sw[i] = ctw[i];
  if (tid < 3) sb[tid] = ctb[tid];
  __syncthreads();

  for (int p = blockIdx.x * blockDim.x + tid; p < cNP; p += gridDim.x * blockDim.x) {
    const int b = p / cHW;
    const int r = p % cHW;
    const int y = r / cW, x = r % cW;
    float l0 = sb[0], l1 = sb[1], l2 = sb[2];
    for (int ci = 0; ci < 67; ++ci) {
      const float* src = (ci < 64) ? (fout + (size_t)(b * cC + ci) * cHW)
                                   : (hns + (size_t)(ci - 64) * cNP + (size_t)b * cHW);
      const int wb = ci * 9;
      #pragma unroll
      for (int ky = 0; ky < 3; ++ky) {
        const int yy = y + ky - 1;
        if (yy < 0 || yy >= cH) continue;
        #pragma unroll
        for (int kx = 0; kx < 3; ++kx) {
          const int xx = x + kx - 1;
          if (xx < 0 || xx >= cW) continue;
          const float v = src[yy * cW + xx];
          const int wi = wb + ky * 3 + kx;
          l0 += sw[wi] * v;
          l1 += sw[603 + wi] * v;
          l2 += sw[1206 + wi] * v;
        }
      }
    }
    const float m = fmaxf(l0, fmaxf(l1, l2));
    const float e0 = __expf(l0 - m), e1 = __expf(l1 - m), e2 = __expf(l2 - m);
    const float is = 1.f / (e0 + e1 + e2);
    out[p] = (e0 * hns[p] + e1 * hns[(size_t)cNP + p] + e2 * hns[(size_t)2 * cNP + p]) * is;
  }
}

// ---------------------------------------------------------------------------
extern "C" void kernel_launch(void* const* d_in, const int* in_sizes, int n_in,
                              void* d_out, int out_size, void* d_ws, size_t ws_size,
                              hipStream_t stream) {
  const float* fout = (const float*)d_in[0];
  const float* hn   = (const float*)d_in[1];
  // d_in[2] (h0) is unused by the reference.
  const float* w3c = (const float*)d_in[3];
  const float* w3g = (const float*)d_in[4];
  const float* w3b = (const float*)d_in[5];
  const float* w5c = (const float*)d_in[6];
  const float* w5g = (const float*)d_in[7];
  const float* w5b = (const float*)d_in[8];
  const float* w7c = (const float*)d_in[9];
  const float* w7g = (const float*)d_in[10];
  const float* w7b = (const float*)d_in[11];
  const float* ckw = (const float*)d_in[12];
  const float* ckb = (const float*)d_in[13];
  const float* ctw = (const float*)d_in[14];
  const float* ctb = (const float*)d_in[15];
  const float* sg3 = (const float*)d_in[16];
  const float* sg5 = (const float*)d_in[17];
  const float* sg7 = (const float*)d_in[18];
  float* out = (float*)d_out;

  // Workspace layout (bytes); all offsets 32B-aligned.
  char* ws = (char*)d_ws;
  const size_t o_in16  = 0;                                        // f16 NHWC fout copy
  const size_t o_afrag = o_in16  + (size_t)cB * cC * cHW * 2;      // packed WMMA A frags
  const size_t o_conv  = o_afrag + (size_t)cMT * cKST * 32 * 16 * 2;
  const size_t o_kbuf  = o_conv  + (size_t)cNCONV * cNP * 4;       // 9+25+49 per-pixel kernels
  const size_t o_conf  = o_kbuf  + (size_t)cNCONV * cNP * 4;
  const size_t o_hn    = o_conf  + (size_t)3 * cNP * 4;            // ping-pong (2 x 3 bufs)
  const size_t o_hns   = o_hn    + (size_t)6 * cNP * 4;
  const size_t o_stat  = o_hns   + (size_t)3 * cNP * 4;            // scale[80], shift[80], part[1280]

  _Float16* in16  = (_Float16*)(ws + o_in16);
  _Float16* afragH= (_Float16*)(ws + o_afrag);
  float* convb = (float*)(ws + o_conv);
  float* kbuf  = (float*)(ws + o_kbuf);
  float* conf  = (float*)(ws + o_conf);
  float* hnbuf = (float*)(ws + o_hn);
  float* hns   = (float*)(ws + o_hns);
  float* scale = (float*)(ws + o_stat);
  float* shift = scale + 80;
  float* part  = shift + 80;

  // 0) fout -> f16 NHWC (channel-innermost) for wide/async staging
  cvt_nhwc<<<dim3(cW / 64, cH, cB), 256, 0, stream>>>(fout, in16);

  // 1) pack weights into A-fragment layout
  prep_afrag<<<(cMT * cKST * 512 + 255) / 256, 256, 0, stream>>>(w3c, w5c, w7c, ckw, afragH);

  // 2) 83-channel 3x3 conv via WMMA implicit GEMM
  conv_wmma<<<dim3(cW / 64, cH, cB), 256, 0, stream>>>(in16, (const v16h*)afragH, convb);

  // 3) batchnorm statistics (deterministic two-stage)
  bn_partial<<<dim3(80, 8), 256, 0, stream>>>(convb, part);
  bn_finalize<<<1, 128, 0, stream>>>(part, w3g, w3b, w5g, w5b, w7g, w7b, scale, shift);

  // 4) per-pixel kernels + conf softmax + hns ch0
  gen_kernels<<<(cNP + 255) / 256, 256, 0, stream>>>(convb, scale, shift, sg3, sg5, sg7,
                                                     ckb, hn, kbuf, conf, hns);

  // 5) six propagation steps
  const float* w3 = kbuf;
  const float* w5 = kbuf + (size_t)9 * cNP;
  const float* w7 = kbuf + (size_t)34 * cNP;
  float* bufA = hnbuf;
  float* bufB = hnbuf + (size_t)3 * cNP;
  const float* s3 = hn; const float* s5 = hn; const float* s7 = hn;
  for (int it = 0; it < 6; ++it) {
    float* dst = (it & 1) ? bufB : bufA;
    const int combo = (it == 2) ? 1 : ((it == 5) ? 2 : -1);
    propagate<<<dim3(cW / 16, cH / 16, cB), 256, 0, stream>>>(
        s3, s5, s7, w3, w5, w7, dst, dst + cNP, dst + 2 * (size_t)cNP, conf, hns, combo);
    s3 = dst; s5 = dst + cNP; s7 = dst + 2 * (size_t)cNP;
  }

  // 6) final ct conv + softmax + blend
  final_ct<<<(cNP + 255) / 256, 256, 0, stream>>>(fout, hns, ctw, ctb, out);
}
